// GaussianVectorQuantizer_9156870275275
// MI455X (gfx1250) — matrix operand order repaired and checked by reading, not compiled, not run on hardware
//
#include <hip/hip_runtime.h>
#include <stdint.h>

// ---------------- problem constants ----------------
#define BATCH   32
#define NPTS    256
#define LATENT  256
#define BOOK    512
#define NCLUS   10
#define TEMP_INV 2.0f      // 1 / TEMPERATURE (0.5)
#define EPS_G   1e-10f

// ---------------- tiling ----------------
#define MT       16        // n-rows per workgroup
#define LC_STRIDE 516      // f32 logits rows: 512 + 4 pad (bank step 4)
#define ZES      264       // bf16 ze rows: 256 + 8 pad  (528B -> bank step 4)
#define ENCS     536       // bf16 encoding rows: 512 + 24 pad (1072B -> bank step 12)

// ws layout
#define WS_BH_OFF 32768                              // bytes; f32 region below
#define BELEMS  (NCLUS * BOOK * LATENT)              // 1,310,720
#define BBYTES  (BELEMS * 2)                         // 2,621,440 per bf16 array

typedef __attribute__((ext_vector_type(8)))  float  v8f;
typedef __attribute__((ext_vector_type(8)))  __bf16 v8bf;
typedef __attribute__((ext_vector_type(16))) __bf16 v16bf;

__device__ __forceinline__ v8f wmma_bf16(v16bf a, v16bf b, v8f c) {
  return __builtin_amdgcn_wmma_f32_16x16x32_bf16(false, a, false, b,
                                                 (short)0, c, false, false);
}
__device__ __forceinline__ v16bf cat8(v8bf a, v8bf b) {
  return __builtin_shufflevector(a, b, 0,1,2,3,4,5,6,7,8,9,10,11,12,13,14,15);
}

// ---------------- counter-based RNG (gumbel), 32-bit ----------------
__device__ __forceinline__ float uniform01(uint32_t x) {
  x ^= x >> 16; x *= 0x7feb352du;
  x ^= x >> 15; x *= 0x846ca68bu;
  x ^= x >> 16;
  return (float)x * 2.3283064365386963e-10f;   // 2^-32
}
__device__ __forceinline__ float gumbelf(uint32_t idx, uint32_t seed) {
  float u = uniform01(idx ^ (seed * 0x9E3779B9u));
  return -__logf(-__logf(u + EPS_G) + EPS_G);
}

// ---------------- kernel A: precisions + cluster gumbel-softmax ----------------
__global__ void gvq_setup(const float* __restrict__ c_logits,
                          const float* __restrict__ lpq,
                          const float* __restrict__ lpqc,
                          float* __restrict__ ws,
                          float* __restrict__ out_precision) {
  int b = threadIdx.x;
  float pq  = 0.5f / fmaxf(1.0f + __expf(lpq[0]),  1e-10f);
  float pqc = 0.5f / fmaxf(1.0f + __expf(lpqc[0]), 1e-10f);
  if (b == 0) { ws[0] = pq; ws[1] = pqc; out_precision[0] = pq; }
  if (b < BATCH) {
    float y[NCLUS];
    float m = -3.0e38f;
    for (int c = 0; c < NCLUS; ++c) {
      float g = gumbelf((uint32_t)(b * NCLUS + c), 0xC0FFEEu);
      float v = (c_logits[b * NCLUS + c] * pqc + g) * TEMP_INV;
      y[c] = v; m = fmaxf(m, v);
    }
    float s = 0.0f;
    for (int c = 0; c < NCLUS; ++c) { y[c] = __expf(y[c] - m); s += y[c]; }
    float inv = 1.0f / s;
    for (int c = 0; c < NCLUS; ++c) ws[2 + b * NCLUS + c] = y[c] * inv;
  }
}

// ---------------- kernel B: per-code squared norms ----------------
__global__ void gvq_booksq(const float* __restrict__ books, float* __restrict__ bsq) {
  int i = blockIdx.x * blockDim.x + threadIdx.x;
  if (i < NCLUS * BOOK) {
    const float4* p = (const float4*)(books + (size_t)i * LATENT);
    float s = 0.0f;
    for (int j = 0; j < LATENT / 4; ++j) {
      float4 v = p[j];
      s += v.x * v.x + v.y * v.y + v.z * v.z + v.w * v.w;
    }
    bsq[i] = s;
  }
}

// ---------------- kernel C: books -> bf16 hi/lo, row-major [c][k][d] ----------------
__global__ void gvq_cvt_row(const float* __restrict__ books,
                            __bf16* __restrict__ bh, __bf16* __restrict__ bl) {
  int t = blockIdx.x * blockDim.x + threadIdx.x;     // one float4 each
  if (t < BELEMS / 4) {
    float4 v = ((const float4*)books)[t];
    float xs[4] = {v.x, v.y, v.z, v.w};
    int base = t * 4;
    for (int j = 0; j < 4; ++j) {
      __bf16 h = (__bf16)xs[j];
      bh[base + j] = h;
      bl[base + j] = (__bf16)(xs[j] - (float)h);
    }
  }
}

// ---------------- kernel D: books -> bf16 hi/lo, transposed [c][d][k] ----------------
__global__ void gvq_cvt_tr(const float* __restrict__ books,
                           __bf16* __restrict__ bth, __bf16* __restrict__ btl) {
  int t = blockIdx.x * blockDim.x + threadIdx.x;     // (c, d, k4), k4 fastest
  if (t < NCLUS * LATENT * (BOOK / 4)) {
    int k4  = t % (BOOK / 4);
    int rem = t / (BOOK / 4);
    int d   = rem % LATENT;
    int c   = rem / LATENT;
    for (int j = 0; j < 4; ++j) {
      int k = k4 * 4 + j;
      float x = books[((size_t)(c * BOOK + k)) * LATENT + d];
      __bf16 h = (__bf16)x;
      size_t o = ((size_t)(c * LATENT + d)) * BOOK + k;
      bth[o] = h;
      btl[o] = (__bf16)(x - (float)h);
    }
  }
}

// ---------------- main fused kernel ----------------
__global__ void __launch_bounds__(256)
gvq_main(const float* __restrict__ ze,
         const float*  __restrict__ wsf,
         const __bf16* __restrict__ bh,  const __bf16* __restrict__ bl,
         const __bf16* __restrict__ bth, const __bf16* __restrict__ btl,
         float* __restrict__ out) {
  extern __shared__ char smem[];
  float*  s_lc   = (float*)smem;                                   // MT*LC_STRIDE f32
  float*  s_la   = s_lc + MT * LC_STRIDE;                          // MT*LC_STRIDE f32
  float*  s_zesq = s_la + MT * LC_STRIDE;                          // 16 f32
  __bf16* s_zeh  = (__bf16*)(s_zesq + 16);                         // MT*ZES
  __bf16* s_zel  = s_zeh + MT * ZES;                               // MT*ZES
  __bf16* s_ench = s_zel + MT * ZES;                               // MT*ENCS
  __bf16* s_encl = s_ench + MT * ENCS;                             // MT*ENCS

  const int tid  = threadIdx.x;
  const int lane = tid & 31;
  const int w    = tid >> 5;
  const int b    = blockIdx.x / (NPTS / MT);
  const int n0   = (blockIdx.x % (NPTS / MT)) * MT;

  const float  pq    = wsf[0];
  const float* cprob = wsf + 2;
  const float* bsq   = wsf + 2 + BATCH * NCLUS;

  // ---- init LDS ----
  if (tid < MT) s_zesq[tid] = 0.0f;
  for (int i = tid; i < MT * LC_STRIDE; i += 256) s_la[i] = 0.0f;
  __syncthreads();

  // ---- stage ze tile: f32 -> bf16 hi/lo + row norms ----
  for (int i = tid; i < MT * (LATENT / 4); i += 256) {
    int row = i / (LATENT / 4), col4 = i % (LATENT / 4);
    float4 v = ((const float4*)(ze + (size_t)(b * NPTS + n0 + row) * LATENT))[col4];
    atomicAdd(&s_zesq[row], v.x * v.x + v.y * v.y + v.z * v.z + v.w * v.w);
    float xs[4] = {v.x, v.y, v.z, v.w};
    int base = row * ZES + col4 * 4;
    for (int j = 0; j < 4; ++j) {
      __bf16 h = (__bf16)xs[j];
      s_zeh[base + j] = h;
      s_zel[base + j] = (__bf16)(xs[j] - (float)h);
    }
  }
  __syncthreads();

  // persistent zq accumulators: this wave owns d-columns [w*32, w*32+32)
  v8f zq0 = {}; v8f zq1 = {};
  const int arow  = lane & 15;
  const int koffA = (lane < 16) ? 0 : 16;   // A: K 0-15 / 16-31
  const int koffB = (lane < 16) ? 0 : 16;   // B: K 0-15 / 16-31
  const int d0a = (w * 2) * 16, d0b = (w * 2 + 1) * 16;

  for (int c = 0; c < NCLUS; ++c) {
    const float cp = cprob[b * NCLUS + c];

    // ---- GEMM1 (split-bf16): logits_c = -(|z|^2 + |e|^2 - 2 z.e) * pq ----
    for (int i = 0; i < 4; ++i) {
      int ct = w * 4 + i, code = ct * 16 + arow;
      const __bf16* ph  = bh + ((size_t)(c * BOOK + code)) * LATENT + koffB;
      const __bf16* pl  = bl + ((size_t)(c * BOOK + code)) * LATENT + koffB;
      const __bf16* pah = s_zeh + arow * ZES + koffA;
      const __bf16* pal = s_zel + arow * ZES + koffA;
      v8f acc = {};
      for (int kk = 0; kk < LATENT; kk += 32) {
        v16bf ah = cat8(*(const v8bf*)(pah + kk), *(const v8bf*)(pah + kk + 8));
        v16bf al = cat8(*(const v8bf*)(pal + kk), *(const v8bf*)(pal + kk + 8));
        v16bf bhv = *(const v16bf*)(ph + kk);
        v16bf blv = *(const v16bf*)(pl + kk);
        acc = wmma_bf16(ah, bhv, acc);
        acc = wmma_bf16(ah, blv, acc);
        acc = wmma_bf16(al, bhv, acc);
      }
      float bq = bsq[c * BOOK + code];
      int col = ct * 16 + arow;
      int rbase = (lane < 16) ? 0 : 8;        // C-layout: VGPR j -> rows j / j+8
      for (int j = 0; j < 8; ++j) {
        float d2 = s_zesq[rbase + j] + bq - 2.0f * acc[j];
        s_lc[(rbase + j) * LC_STRIDE + col] = -d2 * pq;
      }
    }
    __syncthreads();

    // ---- per-row gumbel softmax -> cp*encoding (bf16 hi/lo) + weighted-logit accum ----
    for (int rr = 0; rr < 2; ++rr) {
      int r = w + rr * 8;
      float xv[16], yv[16];
      float m = -3.0e38f;
      for (int i = 0; i < 16; ++i) {
        int col = i * 32 + lane;
        float x = s_lc[r * LC_STRIDE + col];
        uint32_t idx = (uint32_t)((b * NCLUS + c) * NPTS + (n0 + r)) * BOOK + col;
        float y = (x + gumbelf(idx, 0xE5EED5u)) * TEMP_INV;
        xv[i] = x; yv[i] = y;
        m = fmaxf(m, y);
      }
      for (int off = 16; off > 0; off >>= 1) m = fmaxf(m, __shfl_xor(m, off, 32));
      float s = 0.0f;
      for (int i = 0; i < 16; ++i) { yv[i] = __expf(yv[i] - m); s += yv[i]; }
      for (int off = 16; off > 0; off >>= 1) s += __shfl_xor(s, off, 32);
      float inv = cp / s;
      for (int i = 0; i < 16; ++i) {
        int col = i * 32 + lane;
        float e = yv[i] * inv;                 // cp * encoding
        __bf16 h = (__bf16)e;
        s_ench[r * ENCS + col] = h;
        s_encl[r * ENCS + col] = (__bf16)(e - (float)h);
        s_la[r * LC_STRIDE + col] += cp * xv[i];
      }
    }
    __syncthreads();

    // ---- GEMM2 (split-bf16): zq += (cp*encoding)[16x512] @ books_c[512x256] ----
    {
      const __bf16* pah = s_ench + arow * ENCS + koffA;
      const __bf16* pal = s_encl + arow * ENCS + koffA;
      const __bf16* p0h = bth + ((size_t)(c * LATENT + d0a + arow)) * BOOK + koffB;
      const __bf16* p0l = btl + ((size_t)(c * LATENT + d0a + arow)) * BOOK + koffB;
      const __bf16* p1h = bth + ((size_t)(c * LATENT + d0b + arow)) * BOOK + koffB;
      const __bf16* p1l = btl + ((size_t)(c * LATENT + d0b + arow)) * BOOK + koffB;
      for (int kk = 0; kk < BOOK; kk += 32) {
        v16bf ah = cat8(*(const v8bf*)(pah + kk), *(const v8bf*)(pah + kk + 8));
        v16bf al = cat8(*(const v8bf*)(pal + kk), *(const v8bf*)(pal + kk + 8));
        v16bf b0h = *(const v16bf*)(p0h + kk);
        v16bf b0l = *(const v16bf*)(p0l + kk);
        v16bf b1h = *(const v16bf*)(p1h + kk);
        v16bf b1l = *(const v16bf*)(p1l + kk);
        zq0 = wmma_bf16(ah, b0h, zq0);
        zq0 = wmma_bf16(ah, b0l, zq0);
        zq0 = wmma_bf16(al, b0h, zq0);
        zq1 = wmma_bf16(ah, b1h, zq1);
        zq1 = wmma_bf16(ah, b1l, zq1);
        zq1 = wmma_bf16(al, b1h, zq1);
      }
    }
    __syncthreads();   // s_lc / s_enc reused next cluster
  }

  // ---- outputs ----
  const size_t ZQ = (size_t)BATCH * NPTS * LATENT;
  float* out_zq   = out;
  float* out_prob = out + ZQ + 1;
  float* out_logp = out_prob + (size_t)BATCH * NPTS * BOOK;

  for (int rr = 0; rr < 2; ++rr) {
    int r = w + rr * 8;
    float xv[16], ev[16];
    float m = -3.0e38f;
    for (int i = 0; i < 16; ++i) { xv[i] = s_la[r * LC_STRIDE + i * 32 + lane]; m = fmaxf(m, xv[i]); }
    for (int off = 16; off > 0; off >>= 1) m = fmaxf(m, __shfl_xor(m, off, 32));
    float s = 0.0f;
    for (int i = 0; i < 16; ++i) { ev[i] = __expf(xv[i] - m); s += ev[i]; }
    for (int off = 16; off > 0; off >>= 1) s += __shfl_xor(s, off, 32);
    float inv = 1.0f / s, lns = __logf(s);
    size_t rowo = (size_t)(b * NPTS + n0 + r) * BOOK;
    for (int i = 0; i < 16; ++i) {
      int col = i * 32 + lane;
      out_prob[rowo + col] = ev[i] * inv;
      out_logp[rowo + col] = (xv[i] - m) - lns;
    }
  }
  {
    int rbase = (lane < 16) ? 0 : 8;
    for (int j = 0; j < 8; ++j) {
      size_t ro = (size_t)(b * NPTS + n0 + rbase + j) * LATENT;
      out_zq[ro + d0a + arow] = zq0[j];
      out_zq[ro + d0b + arow] = zq1[j];
    }
  }
}

// ---------------- host launcher ----------------
extern "C" void kernel_launch(void* const* d_in, const int* in_sizes, int n_in,
                              void* d_out, int out_size, void* d_ws, size_t ws_size,
                              hipStream_t stream) {
  const float* ze       = (const float*)d_in[0];
  const float* c_logits = (const float*)d_in[1];
  const float* books    = (const float*)d_in[2];
  const float* lpq      = (const float*)d_in[3];
  const float* lpqc     = (const float*)d_in[4];
  float* out = (float*)d_out;
  float* wsf = (float*)d_ws;
  char*  wsb = (char*)d_ws;
  __bf16* bh  = (__bf16*)(wsb + WS_BH_OFF);
  __bf16* bl  = (__bf16*)(wsb + WS_BH_OFF + (size_t)BBYTES);
  __bf16* bth = (__bf16*)(wsb + WS_BH_OFF + (size_t)BBYTES * 2);
  __bf16* btl = (__bf16*)(wsb + WS_BH_OFF + (size_t)BBYTES * 3);

  gvq_setup<<<1, 64, 0, stream>>>(c_logits, lpq, lpqc, wsf,
                                  out + (size_t)BATCH * NPTS * LATENT);
  gvq_booksq<<<(NCLUS * BOOK + 255) / 256, 256, 0, stream>>>(books, wsf + 2 + BATCH * NCLUS);
  gvq_cvt_row<<<(BELEMS / 4 + 255) / 256, 256, 0, stream>>>(books, bh, bl);
  gvq_cvt_tr<<<(NCLUS * LATENT * (BOOK / 4) + 255) / 256, 256, 0, stream>>>(books, bth, btl);

  size_t smem = (size_t)(2 * MT * LC_STRIDE + 16) * sizeof(float)
              + (size_t)(2 * MT * ZES + 2 * MT * ENCS) * sizeof(__bf16);
  hipFuncSetAttribute(reinterpret_cast<const void*>(gvq_main),
                      hipFuncAttributeMaxDynamicSharedMemorySize, (int)smem);
  gvq_main<<<BATCH * (NPTS / MT), 256, smem, stream>>>(ze, wsf, bh, bl, bth, btl, out);
}
